// PerspectiveNet_82386062672602
// MI455X (gfx1250) — matrix operand order repaired and checked by reading, not compiled, not run on hardware
//
#include <hip/hip_runtime.h>
#include <hip/hip_bf16.h>

// ---------------------------------------------------------------------------
// PerspectiveNet (NNUE-style) fused forward for gfx1250 (MI455X, wave32, WMMA)
//
//   hidden = clipsq( [stm;nstm] @ W1^T + b1 )   -> fused, never hits HBM
//   out    = sigmoid( hidden @ W2^T + b2 )
//
// GEMM: M=2*16384 virtual rows, N=1024, K=768, f16 in, f32 WMMA accumulate.
// Each block keeps its 64x768 A-slab resident in LDS (read from HBM once);
// W1 (1.5MB f16) streams from the 192MB L2 through a 3-deep ring of
// GLOBAL_LOAD_ASYNC_TO_LDS_B128 tiles, prefetched 2 iterations ahead so the
// s_wait_asynccnt<=4 at the loop top never waits on a fresh copy.
// ---------------------------------------------------------------------------

typedef __attribute__((ext_vector_type(16))) _Float16 v16h;
typedef __attribute__((ext_vector_type(8)))  _Float16 v8h;
typedef __attribute__((ext_vector_type(4)))  _Float16 v4h;
typedef __attribute__((ext_vector_type(8)))  float    v8f;

#define B_SZ 16384
#define H_SZ 1024
#define F_SZ 768
#define M_SZ (2 * B_SZ)

#define BM  64     // batch rows per block (A slab resident in LDS)
#define BN  128    // hidden units per N-chunk
#define BKS 64     // K columns per staged tile (2 WMMA K-steps of 32)

#define A_STRIDE (F_SZ + 8)   // 776 halves: keeps 16B align, spreads banks
#define B_STRIDE (BKS + 8)    // 72 halves (144B stride, 16B-aligned)
#define LDS_A_HALVES (BM * A_STRIDE)
#define LDS_B_HALVES (BN * B_STRIDE)
#define NBUF 3
#define LDS_BYTES ((LDS_A_HALVES + NBUF * LDS_B_HALVES) * 2)  // ~154.6 KB of 320 KB

// ---- CDNA5 async global->LDS copy (ASYNCcnt-tracked, no VGPR round-trip) ----
__device__ __forceinline__ void async_ld_b128(void* lds_ptr, const void* gptr) {
    // low 32 bits of a generic LDS pointer == wave-relative LDS byte offset
    unsigned int lds_off = (unsigned int)(size_t)lds_ptr;
    asm volatile("global_load_async_to_lds_b128 %0, %1, off"
                 :: "v"(lds_off), "v"(gptr)
                 : "memory");
}
__device__ __forceinline__ void wait_asynccnt0() {
    asm volatile("s_wait_asynccnt 0x0" ::: "memory");
}
__device__ __forceinline__ void wait_asynccnt4() {
    // newest stage group (4 async instrs/wave) may stay in flight;
    // in-order ASYNCcnt completion => everything older has landed
    asm volatile("s_wait_asynccnt 0x4" ::: "memory");
}

__global__ void __launch_bounds__(256)
zero_partial_kernel(float* __restrict__ p) {
    p[blockIdx.x * 256 + threadIdx.x] = 0.0f;
}

// f32 -> f16 convert of W1 (one float4 -> 4 halves per thread)
__global__ void __launch_bounds__(256)
cvt_w1_kernel(const float* __restrict__ W1, _Float16* __restrict__ W1h) {
    int i = blockIdx.x * 256 + threadIdx.x;   // float4 index; grid covers H*F/4
    float4 v = ((const float4*)W1)[i];
    v4h h;
    h[0] = (_Float16)v.x; h[1] = (_Float16)v.y;
    h[2] = (_Float16)v.z; h[3] = (_Float16)v.w;
    ((v4h*)W1h)[i] = h;
}

// stage one BN x BKS f16 tile of W1 into an LDS buffer: 4 async b128 / thread
__device__ __forceinline__ void stage_B_async(_Float16* Bsbuf,
                                              const _Float16* __restrict__ W1h,
                                              int nc, int k0, int tid) {
#pragma unroll
    for (int u = 0; u < 4; ++u) {
        int chunk = u * 256 + tid;       // 1024 x 16B chunks
        int r   = chunk >> 3;            // rows 0..127
        int seg = chunk & 7;             // 8-half segment within BKS=64
        async_ld_b128(Bsbuf + r * B_STRIDE + seg * 8,
                      W1h + (size_t)(nc + r) * F_SZ + k0 + seg * 8);
    }
}

__global__ void __launch_bounds__(256)
nnue_gemm_kernel(const float* __restrict__ stm,
                 const float* __restrict__ nstm,
                 const _Float16* __restrict__ W1h,
                 const float* __restrict__ b1,
                 const float* __restrict__ W2,
                 float* __restrict__ partial) {
    extern __shared__ _Float16 smem[];
    _Float16* As = smem;                   // [BM][A_STRIDE]
    _Float16* B0 = smem + LDS_A_HALVES;    // NBUF x [BN][B_STRIDE] ring

    const int tid    = threadIdx.x;
    const int lane   = tid & 31;
    const int wave   = tid >> 5;
    const int wm     = wave & 1;           // wave row (M): 0..1, 32 rows each
    const int wn     = wave >> 1;          // wave col (N): 0..3, 32 cols each
    const int laneN  = lane & 15;
    const int laneHi = lane >> 4;          // 0 | 1

    const int row0  = blockIdx.x * BM;               // virtual row in [0, 2B)
    const int persp = (row0 >= B_SZ) ? 1 : 0;
    const float* Ag = persp ? nstm : stm;
    const int arow0 = persp ? (row0 - B_SZ) : row0;  // output row base
    const int w2off = persp * H_SZ;                  // W2 half for this persp

    // ---- Phase 1: stage the 64x768 A-slab, f32 -> f16, into LDS ----------
    // 48 float4 per thread, batched 8-deep so 8 global_load_b128 stay in flight
#pragma unroll
    for (int b = 0; b < 6; ++b) {
        float4 v[8];
#pragma unroll
        for (int u = 0; u < 8; ++u) {
            int idx = (b * 8 + u) * 256 + tid;
            int r = idx / 192, c4 = idx % 192;
            v[u] = *(const float4*)(Ag + (size_t)(arow0 + r) * F_SZ + c4 * 4);
        }
#pragma unroll
        for (int u = 0; u < 8; ++u) {
            int idx = (b * 8 + u) * 256 + tid;
            int r = idx / 192, c4 = idx % 192;
            v4h h;
            h[0] = (_Float16)v[u].x; h[1] = (_Float16)v[u].y;
            h[2] = (_Float16)v[u].z; h[3] = (_Float16)v[u].w;
            *(v4h*)(As + r * A_STRIDE + c4 * 4) = h;
        }
    }
    __syncthreads();

    // per-row second-layer dot accumulators, persistent across all N chunks
    float rowsum[2][8];
#pragma unroll
    for (int i = 0; i < 2; ++i)
#pragma unroll
        for (int v = 0; v < 8; ++v) rowsum[i][v] = 0.0f;

    const int kbA = laneHi ? 8 : 0;    // A frag: K {kbA..+7, kbA+16..+23}
    const int kbB = laneHi ? 8 : 0;    // B frag: K {kbB..+7, kbB+16..+23}

    for (int nc = 0; nc < H_SZ; nc += BN) {
        const int nwb = nc + wn * 32;
        float b1v[2], w2v[2];
#pragma unroll
        for (int j = 0; j < 2; ++j) {
            int n = nwb + j * 16 + laneN;
            b1v[j] = b1[n];
            w2v[j] = W2[w2off + n];
        }

        v8f acc[2][2] = {};

        // prologue: two stage groups in flight (prefetch distance 2)
        stage_B_async(B0, W1h, nc, 0, tid);
        stage_B_async(B0 + LDS_B_HALVES, W1h, nc, BKS, tid);

        int it = 0;
        for (int k0 = 0; k0 < F_SZ; k0 += BKS, ++it) {
            // tile k0 must have landed; the distance-2 prefetch may still fly
            if (k0 + BKS < F_SZ) wait_asynccnt4();
            else                 wait_asynccnt0();
            __syncthreads();

            // issue stage(it+2) AFTER the barrier: its target buffer's last
            // readers provably finished before they signaled this barrier
            if (k0 + 2 * BKS < F_SZ)
                stage_B_async(B0 + ((it + 2) % NBUF) * LDS_B_HALVES,
                              W1h, nc, k0 + 2 * BKS, tid);

            const _Float16* Bcur = B0 + (it % NBUF) * LDS_B_HALVES;

#pragma unroll
            for (int ks = 0; ks < BKS; ks += 32) {
                v16h afr[2], bfr[2];
#pragma unroll
                for (int i = 0; i < 2; ++i) {
                    const _Float16* pa = As + (wm * 32 + i * 16 + laneN) * A_STRIDE
                                            + k0 + ks + kbA;
                    v8h lo = *(const v8h*)pa;
                    v8h hi = *(const v8h*)(pa + 16);
                    afr[i] = __builtin_shufflevector(lo, hi, 0, 1, 2, 3, 4, 5, 6, 7,
                                                     8, 9, 10, 11, 12, 13, 14, 15);
                }
#pragma unroll
                for (int j = 0; j < 2; ++j) {
                    const _Float16* pb = Bcur + (wn * 32 + j * 16 + laneN) * B_STRIDE
                                              + ks + kbB;
                    v8h lo = *(const v8h*)pb;
                    v8h hi = *(const v8h*)(pb + 16);
                    bfr[j] = __builtin_shufflevector(lo, hi, 0, 1, 2, 3, 4, 5, 6, 7,
                                                     8, 9, 10, 11, 12, 13, 14, 15);
                }
#pragma unroll
                for (int i = 0; i < 2; ++i)
#pragma unroll
                    for (int j = 0; j < 2; ++j)
                        acc[i][j] = __builtin_amdgcn_wmma_f32_16x16x32_f16(
                            false, afr[i], false, bfr[j], (short)0, acc[i][j],
                            false, false);
            }
        }

        // fused epilogue for this N chunk: +b1, clip [0,1], square, * W2
#pragma unroll
        for (int i = 0; i < 2; ++i)
#pragma unroll
            for (int j = 0; j < 2; ++j)
#pragma unroll
                for (int v = 0; v < 8; ++v) {
                    float h = acc[i][j][v] + b1v[j];
                    h = fminf(fmaxf(h, 0.0f), 1.0f);
                    rowsum[i][v] += h * h * w2v[j];
                }
    }

    // reduce across the 16 N-lanes of each half-wave, then one atomic per row
#pragma unroll
    for (int i = 0; i < 2; ++i) {
#pragma unroll
        for (int v = 0; v < 8; ++v) {
            float s = rowsum[i][v];
            s += __shfl_xor(s, 1, 32);
            s += __shfl_xor(s, 2, 32);
            s += __shfl_xor(s, 4, 32);
            s += __shfl_xor(s, 8, 32);
            rowsum[i][v] = s;
        }
        if (laneN == 0) {
            // D layout: VGPR v -> M=v (lanes 0-15), M=8+v (lanes 16-31)
            int mbase = wm * 32 + i * 16 + (laneHi ? 8 : 0);
#pragma unroll
            for (int v = 0; v < 8; ++v)
                atomicAdd(partial + arow0 + mbase + v, rowsum[i][v]);
        }
    }
}

__global__ void __launch_bounds__(256)
finalize_kernel(const float* __restrict__ partial,
                const float* __restrict__ b2,
                float* __restrict__ out) {
    int i = blockIdx.x * 256 + threadIdx.x;
    float x = partial[i] + b2[0];
    out[i] = 1.0f / (1.0f + __expf(-x));
}

extern "C" void kernel_launch(void* const* d_in, const int* in_sizes, int n_in,
                              void* d_out, int out_size, void* d_ws, size_t ws_size,
                              hipStream_t stream) {
    const float* stm  = (const float*)d_in[0];
    const float* nstm = (const float*)d_in[1];
    const float* W1   = (const float*)d_in[2];
    const float* b1   = (const float*)d_in[3];
    const float* W2   = (const float*)d_in[4];
    const float* b2   = (const float*)d_in[5];
    float* out = (float*)d_out;

    // workspace layout: [16384 f32 partial][1024*768 f16 W1h]
    float*    partial = (float*)d_ws;
    _Float16* W1h     = (_Float16*)((char*)d_ws + (size_t)B_SZ * sizeof(float));

    zero_partial_kernel<<<B_SZ / 256, 256, 0, stream>>>(partial);
    cvt_w1_kernel<<<(H_SZ * F_SZ / 4) / 256, 256, 0, stream>>>(W1, W1h);
    nnue_gemm_kernel<<<M_SZ / BM, 256, LDS_BYTES, stream>>>(stm, nstm, W1h, b1, W2, partial);
    finalize_kernel<<<B_SZ / 256, 256, 0, stream>>>(partial, b2, out);
}
